// NonLocalBlockND_34394098106431
// MI455X (gfx1250) — matrix operand order repaired and checked by reading, not compiled, not run on hardware
//
#include <hip/hip_runtime.h>

// ---------------- types & helpers ----------------
typedef __attribute__((ext_vector_type(16))) __bf16 v16bf;
typedef __attribute__((ext_vector_type(8)))  __bf16 v8bf;
typedef __attribute__((ext_vector_type(4)))  __bf16 v4bf;
typedef __attribute__((ext_vector_type(8)))  float  v8f;
typedef __attribute__((ext_vector_type(4)))  float  v4f;

// native converts -> v_cvt_pk_bf16_f32 class ops instead of manual bit math
__device__ __forceinline__ __bf16 f2bf(float f)  { return (__bf16)f; }
__device__ __forceinline__ float  bf2f(__bf16 h) { return (float)h; }

__device__ __forceinline__ v16bf cat8(v8bf lo, v8bf hi) {
  return __builtin_shufflevector(lo, hi, 0,1,2,3,4,5,6,7,8,9,10,11,12,13,14,15);
}

#define USE_ASYNC_LDS 1

// ---------------- problem dims ----------------
constexpr int Bb = 4, C = 256, CI = 128, T = 8, H = 32, Wd = 32;
constexpr int Nn = T * H * Wd;          // 8192
constexpr int NK = T * (H/2) * (Wd/2);  // 2048

// =====================================================================
// Kernel 1: the three 1x1 input projections as WMMA GEMMs.
// =====================================================================
__global__ __launch_bounds__(256) void proj_kernel(
    const float* __restrict__ x,
    const float* __restrict__ gw, const float* __restrict__ gb,
    const float* __restrict__ tw, const float* __restrict__ tb,
    const float* __restrict__ pw, const float* __restrict__ pb,
    __bf16* __restrict__ out_g, __bf16* __restrict__ out_t, __bf16* __restrict__ out_p)
{
  __shared__ __bf16 sX[16 * C];               // [n_local][c]
  int wg    = blockIdx.x;
  int batch = wg / (Nn / 16);
  int n0    = (wg % (Nn / 16)) * 16;
  int tid   = threadIdx.x;

  { // cooperative load + convert of x tile: pack 2 adjacent channels -> b32 LDS store
    int nl = tid % 16;
    int cp = tid / 16;                        // 0..15 channel-pair group
    const float* xb = x + ((size_t)batch * C) * Nn + n0 + nl;
    #pragma unroll
    for (int it = 0; it < 8; ++it) {
      int c = 2 * (cp + 16 * it);
      __bf16 b0 = f2bf(xb[(size_t)c * Nn]);
      __bf16 b1 = f2bf(xb[(size_t)(c + 1) * Nn]);
      unsigned w = (unsigned)__builtin_bit_cast(unsigned short, b0) |
                   ((unsigned)__builtin_bit_cast(unsigned short, b1) << 16);
      *(unsigned*)&sX[nl * C + c] = w;
    }
  }
  __syncthreads();

  int wave = tid / 32, lane = tid % 32;
  int half = lane >> 4, l16 = lane & 15;
  int o0 = wave * 16;

  const float* wptr[3] = {gw, tw, pw};
  const float* bptr[3] = {gb, tb, pb};
  __bf16* optr[3];
  optr[0] = out_g + ((size_t)batch * Nn + n0) * CI;
  optr[1] = out_t + ((size_t)batch * Nn + n0) * CI;
  optr[2] = out_p + ((size_t)batch * Nn + n0) * CI;

  #pragma unroll
  for (int p = 0; p < 3; ++p) {
    const float* Wm = wptr[p];
    v8f acc = {};
    #pragma unroll
    for (int kk = 0; kk < C / 32; ++kk) {
      // A fragment (weights): lane -> row o0+l16; K contiguous per half
      int kg0 = kk * 32 + half * 16;
      const float* wr = Wm + (size_t)(o0 + l16) * C;
      v8f flo = *(const v8f*)(wr + kg0);
      v8f fhi = *(const v8f*)(wr + kg0 + 8);
      v16bf a = cat8(__builtin_convertvector(flo, v8bf),
                     __builtin_convertvector(fhi, v8bf));
      // B fragment (x tile): lane -> col n=l16; K contiguous per half
      v16bf bfrag = *(const v16bf*)(&sX[l16 * C + kk * 32 + half * 16]);
      acc = __builtin_amdgcn_wmma_f32_16x16x32_bf16(
                false, a, false, bfrag, (short)0, acc, false, false);
    }
    const float* bias = bptr[p];
    v8bf st;
    #pragma unroll
    for (int r = 0; r < 8; ++r) st[r] = f2bf(acc[r] + bias[o0 + half * 8 + r]);
    *(v8bf*)(optr[p] + (size_t)l16 * CI + o0 + half * 8) = st;
  }
}

// =====================================================================
// Kernel 2: MaxPool3d((1,2,2)) on [b][n][ci] bf16 -> [b][nk][ci]
// =====================================================================
__global__ __launch_bounds__(256) void pool_kernel(
    const __bf16* __restrict__ inp, __bf16* __restrict__ outp)
{
  int g = blockIdx.x * blockDim.x + threadIdx.x;
  int total = Bb * NK * (CI / 4);
  if (g >= total) return;
  int bk  = g / (NK * (CI / 4));
  int rem = g % (NK * (CI / 4));
  int nk  = rem / (CI / 4);
  int cq  = (rem % (CI / 4)) * 4;
  int t = nk / 256, hp = (nk / 16) % 16, wp = nk % 16;
  int nb = t * 1024 + hp * 64 + wp * 2;
  const __bf16* base = inp + ((size_t)bk * Nn) * CI + cq;
  int ns[4] = {nb, nb + 1, nb + 32, nb + 33};
  float  m[4];
  __bf16 mv[4];
  #pragma unroll
  for (int s = 0; s < 4; ++s) {
    v4bf rv = *(const v4bf*)(base + (size_t)ns[s] * CI);
    #pragma unroll
    for (int j = 0; j < 4; ++j) {
      float v = bf2f(rv[j]);
      if (s == 0 || v > m[j]) { m[j] = v; mv[j] = rv[j]; }
    }
  }
  __bf16* o = outp + ((size_t)bk * NK + nk) * CI + cq;
  v4bf ov = {mv[0], mv[1], mv[2], mv[3]};
  *(v4bf*)o = ov;
}

// =====================================================================
// Kernel 3: fused flash attention (f never materialized).
// phi chunk staged with GLOBAL_LOAD_ASYNC_TO_LDS_B128 (ASYNCcnt path),
// g chunk transposed with packed b32 DS stores.
// =====================================================================
__global__ __launch_bounds__(256) void attn_kernel(
    const __bf16* __restrict__ theta,   // [B][N][CI]
    const __bf16* __restrict__ phiT,    // [B][NK][CI]
    const __bf16* __restrict__ gT,      // [B][NK][CI]
    __bf16* __restrict__ yout)          // [B][N][CI]
{
  __shared__ __bf16 sPhi[32 * CI];      // [nk_local][ci]
  __shared__ __bf16 sGt[CI * 32];       // [ci][nk_local]
  __shared__ __bf16 sP[8][16 * 32];     // per-wave P scratch

  int wgPerBatch = Nn / 128;
  int batch = blockIdx.x / wgPerBatch;
  int nblk  = (blockIdx.x % wgPerBatch) * 128;
  int tid = threadIdx.x, wave = tid / 32, lane = tid % 32;
  int half = lane >> 4, l16 = lane & 15;
  int n0 = nblk + wave * 16;

  // theta A-fragments (16 query rows, K = 128)
  v16bf a[4];
  {
    const __bf16* tr = theta + ((size_t)batch * Nn + n0 + l16) * CI;
    #pragma unroll
    for (int kk = 0; kk < 4; ++kk) {
      int kg = kk * 32 + half * 16;
      a[kk] = cat8(*(const v8bf*)(tr + kg), *(const v8bf*)(tr + kg + 8));
    }
  }

  float mrow[8], lrow[8];
  #pragma unroll
  for (int r = 0; r < 8; ++r) { mrow[r] = -1e30f; lrow[r] = 0.f; }
  v8f yacc[8] = {};

  const __bf16* phiB = phiT + (size_t)batch * NK * CI;
  const __bf16* gB   = gT   + (size_t)batch * NK * CI;

  // staging thread mappings
  int prow = tid / 8;                 // 0..31  (phi row)
  int pseg = (tid % 8) * 16;          // phi ci offset (2x b128 per thread)
  int grp  = tid / 16;                // 0..15  (g row-pair)
  int gseg = (tid % 16) * 8;          // g ci offset (8 ci per thread)

  for (int ch = 0; ch < NK; ch += 32) {
    __syncthreads();     // previous chunk fully consumed
    {
      // ---- phi chunk: async DMA global -> LDS (no VGPR round trip) ----
      const __bf16* psrc = phiB + (size_t)(ch + prow) * CI + pseg;
#if USE_ASYNC_LDS
      unsigned ldst = (unsigned)(uintptr_t)(void*)&sPhi[prow * CI + pseg];
      unsigned long long ga = (unsigned long long)(uintptr_t)(const void*)psrc;
      asm volatile("global_load_async_to_lds_b128 %0, %1, off"
                   :: "v"(ldst), "v"(ga) : "memory");
      asm volatile("global_load_async_to_lds_b128 %0, %1, off offset:16"
                   :: "v"(ldst), "v"(ga) : "memory");
#else
      *(v8bf*)(&sPhi[prow * CI + pseg])     = *(const v8bf*)(psrc);
      *(v8bf*)(&sPhi[prow * CI + pseg + 8]) = *(const v8bf*)(psrc + 8);
#endif
      // ---- g chunk: transpose, 2 adjacent rows packed into b32 stores ----
      const __bf16* g0 = gB + (size_t)(ch + 2 * grp) * CI + gseg;
      v8bf r0 = *(const v8bf*)(g0);
      v8bf r1 = *(const v8bf*)(g0 + CI);
      #pragma unroll
      for (int e = 0; e < 8; ++e) {
        unsigned w = (unsigned)__builtin_bit_cast(unsigned short, r0[e]) |
                     ((unsigned)__builtin_bit_cast(unsigned short, r1[e]) << 16);
        *(unsigned*)&sGt[(gseg + e) * 32 + 2 * grp] = w;
      }
      // prefetch next chunk into cache while we compute on this one
      if (ch + 32 < NK) {
        __builtin_prefetch(phiB + (size_t)(ch + 32 + prow) * CI + pseg, 0, 1);
        __builtin_prefetch(gB   + (size_t)(ch + 32 + 2 * grp) * CI + gseg, 0, 1);
      }
    }
#if USE_ASYNC_LDS
    asm volatile("s_wait_asynccnt 0x0" ::: "memory");
#endif
    __syncthreads();

    // S = theta_tile @ phi_chunk^T  -> two 16x16 f32 tiles
    v8f s0 = {}, s1 = {};
    #pragma unroll
    for (int kk = 0; kk < 4; ++kk) {
      v16bf b0 = *(const v16bf*)(&sPhi[l16 * CI + kk * 32 + half * 16]);
      v16bf b1 = *(const v16bf*)(&sPhi[(16 + l16) * CI + kk * 32 + half * 16]);
      s0 = __builtin_amdgcn_wmma_f32_16x16x32_bf16(false, a[kk], false, b0, (short)0, s0, false, false);
      s1 = __builtin_amdgcn_wmma_f32_16x16x32_bf16(false, a[kk], false, b1, (short)0, s1, false, false);
    }

    // online softmax per query row (row = half*8 + r, cols across 16 lanes)
    float scale[8];
    __bf16* Pw = &sP[wave][0];
    #pragma unroll
    for (int r = 0; r < 8; ++r) {
      float mx = fmaxf(s0[r], s1[r]);
      #pragma unroll
      for (int d = 1; d < 16; d <<= 1) mx = fmaxf(mx, __shfl_xor(mx, d, 16));
      float mnew = fmaxf(mrow[r], mx);
      float sc = __expf(mrow[r] - mnew);
      float p0 = __expf(s0[r] - mnew);
      float p1 = __expf(s1[r] - mnew);
      float rs = p0 + p1;
      #pragma unroll
      for (int d = 1; d < 16; d <<= 1) rs += __shfl_xor(rs, d, 16);
      lrow[r] = lrow[r] * sc + rs;
      mrow[r] = mnew;
      scale[r] = sc;
      int pr = half * 8 + r;
      unsigned w = (unsigned)__builtin_bit_cast(unsigned short, f2bf(p0)) |
                   ((unsigned)__builtin_bit_cast(unsigned short, f2bf(p1)) << 16);
      // cols l16 and 16+l16 are 32B apart; write both halves separately
      Pw[pr * 32 + l16]      = __builtin_bit_cast(__bf16, (unsigned short)(w & 0xffff));
      Pw[pr * 32 + 16 + l16] = __builtin_bit_cast(__bf16, (unsigned short)(w >> 16));
    }
    #pragma unroll
    for (int j = 0; j < 8; ++j)
      #pragma unroll
      for (int r = 0; r < 8; ++r) yacc[j][r] *= scale[r];

    // reload P in A-fragment layout (wave-private LDS; DS in-order per wave)
    const __bf16* prd = &Pw[l16 * 32];
    v16bf pa = cat8(*(const v8bf*)(prd + half * 16),
                    *(const v8bf*)(prd + half * 16 + 8));

    // y += P @ g_chunk   (8 column tiles of ci)
    #pragma unroll
    for (int j = 0; j < 8; ++j) {
      v16bf bg = *(const v16bf*)(&sGt[(j * 16 + l16) * 32 + half * 16]);
      yacc[j] = __builtin_amdgcn_wmma_f32_16x16x32_bf16(false, pa, false, bg, (short)0, yacc[j], false, false);
    }
  }

  // normalize by row sums and store y (bf16, [n][ci])
  float inv[8];
  #pragma unroll
  for (int r = 0; r < 8; ++r) inv[r] = 1.0f / lrow[r];
  __bf16* yb = yout + ((size_t)batch * Nn) * CI;
  #pragma unroll
  for (int j = 0; j < 8; ++j) {
    int ci = j * 16 + l16;
    #pragma unroll
    for (int r = 0; r < 8; ++r)
      yb[(size_t)(n0 + half * 8 + r) * CI + ci] = f2bf(yacc[j][r] * inv[r]);
  }
}

// =====================================================================
// Kernel 4: W projection (128 -> 256) + per-channel sum / sum^2 for BN.
// =====================================================================
__global__ __launch_bounds__(256) void wconv_kernel(
    const __bf16* __restrict__ y,       // [B][N][CI]
    const float* __restrict__ Ww,       // [C][CI]
    const float* __restrict__ Wb,
    float* __restrict__ z,              // [B][C][N]
    float* __restrict__ gsum, float* __restrict__ gsqs)
{
  __shared__ __bf16 sW[128 * CI];       // 32 KB, one o-half at a time
  __shared__ float  sSum[C], sSqs[C];
  int tid = threadIdx.x;
  for (int i = tid; i < C; i += 256) { sSum[i] = 0.f; sSqs[i] = 0.f; }

  int wgPerBatch = Nn / 128;
  int batch = blockIdx.x / wgPerBatch;
  int nblk  = (blockIdx.x % wgPerBatch) * 128;
  int wave = tid / 32, lane = tid % 32, half = lane >> 4, l16 = lane & 15;
  int n0 = nblk + wave * 16;

  // A fragments from y (16 rows x K=128)
  v16bf a[4];
  const __bf16* yr = y + ((size_t)batch * Nn + n0 + l16) * CI;
  #pragma unroll
  for (int kk = 0; kk < 4; ++kk) {
    int kg = kk * 32 + half * 16;
    a[kk] = cat8(*(const v8bf*)(yr + kg), *(const v8bf*)(yr + kg + 8));
  }

  float* zb = z + ((size_t)batch * C) * Nn;
  for (int hfo = 0; hfo < 2; ++hfo) {
    __syncthreads();
    for (int i = tid * 4; i < 128 * CI; i += 256 * 4) {
      v4f v = *(const v4f*)(Ww + (size_t)hfo * 128 * CI + i);
      *(v4bf*)&sW[i] = __builtin_convertvector(v, v4bf);
    }
    __syncthreads();
    #pragma unroll
    for (int j = 0; j < 8; ++j) {
      int ol = j * 16 + l16;
      int o  = hfo * 128 + ol;
      v8f acc = {};
      #pragma unroll
      for (int kk = 0; kk < 4; ++kk) {
        v16bf bfrag = *(const v16bf*)(&sW[ol * CI + kk * 32 + half * 16]);
        acc = __builtin_amdgcn_wmma_f32_16x16x32_bf16(false, a[kk], false, bfrag, (short)0, acc, false, false);
      }
      float bias = Wb[o];
      float ps = 0.f, pq = 0.f;
      #pragma unroll
      for (int r = 0; r < 8; ++r) { acc[r] += bias; ps += acc[r]; pq += acc[r] * acc[r]; }
      float* zp = zb + (size_t)o * Nn + n0 + half * 8;
      *(v4f*)zp       = (v4f){acc[0], acc[1], acc[2], acc[3]};
      *(v4f*)(zp + 4) = (v4f){acc[4], acc[5], acc[6], acc[7]};
      atomicAdd(&sSum[o], ps);
      atomicAdd(&sSqs[o], pq);
    }
  }
  __syncthreads();
  for (int i = tid; i < C; i += 256) {
    atomicAdd(&gsum[i], sSum[i]);
    atomicAdd(&gsqs[i], sSqs[i]);
  }
}

// =====================================================================
// Kernel 5: batch-norm (training-mode stats) + gamma/beta + residual.
// =====================================================================
__global__ __launch_bounds__(256) void bn_kernel(
    const float* __restrict__ z, const float* __restrict__ x,
    const float* __restrict__ gsum, const float* __restrict__ gsqs,
    const float* __restrict__ gamma, const float* __restrict__ beta,
    float* __restrict__ out)
{
  size_t i = (size_t)blockIdx.x * blockDim.x + threadIdx.x;
  size_t total = (size_t)Bb * C * Nn;
  if (i >= total) return;
  int o = (int)((i / Nn) % C);
  float cnt  = (float)(Bb * Nn);
  float mean = gsum[o] / cnt;
  float var  = gsqs[o] / cnt - mean * mean;
  float is   = rsqrtf(var + 1e-5f);
  out[i] = (z[i] - mean) * is * gamma[o] + beta[o] + x[i];
}

// =====================================================================
extern "C" void kernel_launch(void* const* d_in, const int* in_sizes, int n_in,
                              void* d_out, int out_size, void* d_ws, size_t ws_size,
                              hipStream_t stream) {
  const float* x   = (const float*)d_in[0];
  const float* g_w = (const float*)d_in[1];
  const float* g_b = (const float*)d_in[2];
  const float* t_w = (const float*)d_in[3];
  const float* t_b = (const float*)d_in[4];
  const float* p_w = (const float*)d_in[5];
  const float* p_b = (const float*)d_in[6];
  const float* W_w = (const float*)d_in[7];
  const float* W_b = (const float*)d_in[8];
  const float* bng = (const float*)d_in[9];
  const float* bnb = (const float*)d_in[10];
  float* out = (float*)d_out;

  char* ws = (char*)d_ws;
  const size_t SZ_FULL = (size_t)Bb * Nn * CI * sizeof(__bf16);   // 8 MB
  const size_t SZ_POOL = (size_t)Bb * NK * CI * sizeof(__bf16);   // 2 MB
  __bf16* theta    = (__bf16*)(ws);
  __bf16* phi_full = (__bf16*)(ws + SZ_FULL);
  __bf16* g_full   = (__bf16*)(ws + 2 * SZ_FULL);
  __bf16* phiT     = (__bf16*)(ws + 3 * SZ_FULL);
  __bf16* gTp      = (__bf16*)(ws + 3 * SZ_FULL + SZ_POOL);
  __bf16* ybuf     = (__bf16*)(ws + 3 * SZ_FULL + 2 * SZ_POOL);
  float*  zbuf     = (float*) (ws + 4 * SZ_FULL + 2 * SZ_POOL);
  float*  gsum     = (float*) (ws + 4 * SZ_FULL + 2 * SZ_POOL + (size_t)Bb * C * Nn * 4);
  float*  gsqs     = gsum + C;

  proj_kernel<<<Bb * (Nn / 16), 256, 0, stream>>>(
      x, g_w, g_b, t_w, t_b, p_w, p_b, g_full, theta, phi_full);

  int poolBlocks = (Bb * NK * (CI / 4) + 255) / 256;
  pool_kernel<<<poolBlocks, 256, 0, stream>>>(phi_full, phiT);
  pool_kernel<<<poolBlocks, 256, 0, stream>>>(g_full, gTp);

  attn_kernel<<<Bb * (Nn / 128), 256, 0, stream>>>(theta, phiT, gTp, ybuf);

  hipMemsetAsync(gsum, 0, 2 * C * sizeof(float), stream);
  wconv_kernel<<<Bb * (Nn / 128), 256, 0, stream>>>(ybuf, W_w, W_b, zbuf, gsum, gsqs);

  size_t total = (size_t)Bb * C * Nn;
  bn_kernel<<<(unsigned)((total + 255) / 256), 256, 0, stream>>>(
      zbuf, x, gsum, gsqs, bng, bnb, out);
}
